// QADebiasedContrastiveLoss_33354716021154
// MI455X (gfx1250) — matrix operand order repaired and checked by reading, not compiled, not run on hardware
//
#include <hip/hip_runtime.h>
#include <hip/hip_bf16.h>
#include <math.h>

// MI455X / gfx1250, wave32. Split-f16 WMMA GEMM (hi+lo) for ~fp32 accuracy.
// A (question) fragments are explicitly register-resident across the column
// loop; B (sentence) fragments stream from global (L1/L2-resident).

typedef _Float16 half_t;
typedef __attribute__((ext_vector_type(8)))  _Float16 v8h;
typedef __attribute__((ext_vector_type(16))) _Float16 v16h;
typedef __attribute__((ext_vector_type(8)))  float    v8f;

#define DDIM 256
#define SDIM 64
#define QDIM 4096
#define NDIM 16384
#define NSLICE 16
#define GRIDB ((QDIM / 128) * NSLICE)   // 512 pass-B partials
#define TAU_F 0.05f
#define KCH (DDIM / 32)                 // 8 K-chunks of 32

__device__ __forceinline__ v16h cat8(v8h a, v8h b) {
  return __builtin_shufflevector(a, b, 0,1,2,3,4,5,6,7,8,9,10,11,12,13,14,15);
}

// Preload this lane's A row (row = q0+idx) as 8 hi + 8 lo v16h fragments.
// Per ISA 16-bit A layout: lane half h needs K in [k0+8h,+8) U [k0+16+8h,+8).
__device__ __forceinline__ void load_a_frags(
    const half_t* __restrict__ ahi, const half_t* __restrict__ alo,
    int h, v16h* aH, v16h* aL) {
#pragma unroll
  for (int kc = 0; kc < KCH; ++kc) {
    int k0 = kc * 32;
    aH[kc] = cat8(*(const v8h*)(ahi + k0 + 8 * h),
                  *(const v8h*)(ahi + k0 + 16 + 8 * h));
    aL[kc] = cat8(*(const v8h*)(alo + k0 + 8 * h),
                  *(const v8h*)(alo + k0 + 16 + 8 * h));
  }
}

// 16x16 tile with register-resident A; B row streamed (3 WMMAs per K=32:
// aH*bH + aH*bL + aL*bH recovers ~fp32 accuracy).
__device__ __forceinline__ v8f tile_from_regs(
    const v16h* aH, const v16h* aL,
    const half_t* __restrict__ bhi, const half_t* __restrict__ blo, int h) {
  v8f c = {};
#pragma unroll
  for (int kc = 0; kc < KCH; ++kc) {
    int k0 = kc * 32;
    v16h bH = cat8(*(const v8h*)(bhi + k0 + 8 * h), *(const v8h*)(bhi + k0 + 16 + 8 * h));
    v16h bL = cat8(*(const v8h*)(blo + k0 + 8 * h), *(const v8h*)(blo + k0 + 16 + 8 * h));
    c = __builtin_amdgcn_wmma_f32_16x16x32_f16(false, aH[kc], false, bH, (short)0, c, false, false);
    c = __builtin_amdgcn_wmma_f32_16x16x32_f16(false, aH[kc], false, bL, (short)0, c, false, false);
    c = __builtin_amdgcn_wmma_f32_16x16x32_f16(false, aL[kc], false, bH, (short)0, c, false, false);
  }
  return c;
}

// Load-in-loop variant for the tiny SQ GEMM (one tile per wave).
__device__ __forceinline__ v8f tile_gemm(
    const half_t* __restrict__ ahi, const half_t* __restrict__ alo,
    const half_t* __restrict__ bhi, const half_t* __restrict__ blo, int h) {
  v16h aH[KCH], aL[KCH];
  load_a_frags(ahi, alo, h, aH, aL);
  return tile_from_regs(aH, aL, bhi, blo, h);
}

__device__ __forceinline__ float block_reduce_sum(float v) {
  __shared__ float red[256];
  int t = threadIdx.x;
  red[t] = v; __syncthreads();
  for (int s = 128; s > 0; s >>= 1) { if (t < s) red[t] += red[t + s]; __syncthreads(); }
  float r = red[0]; __syncthreads();
  return r;
}

// Normalize rows of [rows, 256] and emit hi/lo f16 split. blockDim == 256.
__global__ void k_normalize_split(const float* __restrict__ in,
                                  half_t* __restrict__ hi, half_t* __restrict__ lo) {
  __shared__ float red[DDIM];
  int row = blockIdx.x, t = threadIdx.x;
  float x = in[(size_t)row * DDIM + t];
  red[t] = x * x; __syncthreads();
  for (int s = 128; s > 0; s >>= 1) { if (t < s) red[t] += red[t + s]; __syncthreads(); }
  float y = x / sqrtf(red[0]);
  half_t hv = (half_t)y;
  hi[(size_t)row * DDIM + t] = hv;
  lo[(size_t)row * DDIM + t] = (half_t)(y - (float)hv);
}

// Pass A: per-question sum(exp), sum_pos(exp), count over one N-slice.
__global__ void __launch_bounds__(256, 1)
k_pass_a(const half_t* __restrict__ quehi, const half_t* __restrict__ quelo,
         const half_t* __restrict__ senhi, const half_t* __restrict__ senlo,
         const int* __restrict__ parq, const int* __restrict__ senp,
         float* __restrict__ pa) {
  int lane = threadIdx.x & 31, wave = threadIdx.x >> 5;
  int idx = lane & 15, h = lane >> 4;
  int q0 = (blockIdx.x * 8 + wave) * 16;
  v16h aH[KCH], aL[KCH];
  load_a_frags(quehi + (size_t)(q0 + idx) * DDIM,
               quelo + (size_t)(q0 + idx) * DDIM, h, aH, aL);
  int pid[8];
#pragma unroll
  for (int r = 0; r < 8; ++r) pid[r] = parq[q0 + r + 8 * h];
  float accA[8] = {0}, accP[8] = {0}, accC[8] = {0};
  int nBeg = blockIdx.y * (NDIM / NSLICE);
  int nEnd = nBeg + (NDIM / NSLICE);
  for (int n0 = nBeg; n0 < nEnd; n0 += 16) {
    const half_t* bh = senhi + (size_t)(n0 + idx) * DDIM;
    const half_t* bl = senlo + (size_t)(n0 + idx) * DDIM;
    int sp = senp[n0 + idx];
    v8f c = tile_from_regs(aH, aL, bh, bl, h);
#pragma unroll
    for (int r = 0; r < 8; ++r) {
      float s = c[r] / TAU_F;
      float e = expf(s);
      bool p = (sp == pid[r]);
      accA[r] += e;
      accP[r] += p ? e : 0.0f;
      accC[r] += p ? 1.0f : 0.0f;
    }
  }
#pragma unroll
  for (int r = 0; r < 8; ++r) {
    float va = accA[r], vp = accP[r], vc = accC[r];
    for (int m = 1; m < 16; m <<= 1) {
      va += __shfl_xor(va, m, 32);
      vp += __shfl_xor(vp, m, 32);
      vc += __shfl_xor(vc, m, 32);
    }
    if (idx == 0) {
      int q = q0 + r + 8 * h;
      float* p = pa + ((size_t)blockIdx.y * QDIM + q) * 3;
      p[0] = va; p[1] = vp; p[2] = vc;
    }
  }
}

// Per-question debiased estimator g -> Nn*g, validity, masked M.
__global__ void k_g(const float* __restrict__ pa, const int* __restrict__ pnum,
                    float* __restrict__ nng, float* __restrict__ vflag,
                    float* __restrict__ mval) {
  int q = blockIdx.x * blockDim.x + threadIdx.x;
  if (q >= QDIM) return;
  float sa = 0.f, sp = 0.f, sc = 0.f;
  for (int s = 0; s < NSLICE; ++s) {
    const float* p = pa + ((size_t)s * QDIM + q) * 3;
    sa += p[0]; sp += p[1]; sc += p[2];
  }
  int M = (int)(sc + 0.5f);
  int Nn = NDIM - M;
  int P = pnum[0] > 1 ? pnum[0] : 1;
  float eta_p = 1.0f / (float)P;
  float eta_m = 1.0f - eta_p;
  float sneg = sa - sp;
  float Mf = (float)(M > 1 ? M : 1);
  float Nf = (float)(Nn > 1 ? Nn : 1);
  float minv = expf(-1.0f / TAU_F);
  float g = fmaxf((sneg / Nf - eta_p * sp / Mf) / eta_m, minv);
  bool valid = (M > 0) && (Nn > 0);
  nng[q] = (float)Nn * g;
  vflag[q] = valid ? 1.0f : 0.0f;
  mval[q] = valid ? (float)M : 0.0f;
}

// Pass B: redo GEMM, accumulate loss at positive entries: log(e^s + Nn*g) - s.
__global__ void __launch_bounds__(256, 1)
k_pass_b(const half_t* __restrict__ quehi, const half_t* __restrict__ quelo,
         const half_t* __restrict__ senhi, const half_t* __restrict__ senlo,
         const int* __restrict__ parq, const int* __restrict__ senp,
         const float* __restrict__ nng, const float* __restrict__ vflag,
         float* __restrict__ lossB) {
  __shared__ float lpart[8];
  int lane = threadIdx.x & 31, wave = threadIdx.x >> 5;
  int idx = lane & 15, h = lane >> 4;
  int q0 = (blockIdx.x * 8 + wave) * 16;
  v16h aH[KCH], aL[KCH];
  load_a_frags(quehi + (size_t)(q0 + idx) * DDIM,
               quelo + (size_t)(q0 + idx) * DDIM, h, aH, aL);
  int pid[8]; float ngr[8], vfr[8];
#pragma unroll
  for (int r = 0; r < 8; ++r) {
    int q = q0 + r + 8 * h;
    pid[r] = parq[q]; ngr[r] = nng[q]; vfr[r] = vflag[q];
  }
  float acc = 0.0f;
  int nBeg = blockIdx.y * (NDIM / NSLICE);
  int nEnd = nBeg + (NDIM / NSLICE);
  for (int n0 = nBeg; n0 < nEnd; n0 += 16) {
    const half_t* bh = senhi + (size_t)(n0 + idx) * DDIM;
    const half_t* bl = senlo + (size_t)(n0 + idx) * DDIM;
    int sp = senp[n0 + idx];
    v8f c = tile_from_regs(aH, aL, bh, bl, h);
#pragma unroll
    for (int r = 0; r < 8; ++r) {
      if (vfr[r] != 0.0f && sp == pid[r]) {
        float s = c[r] / TAU_F;
        acc += logf(expf(s) + ngr[r]) - s;
      }
    }
  }
  for (int m = 1; m < 32; m <<= 1) acc += __shfl_xor(acc, m, 32);
  if (lane == 0) lpart[wave] = acc;
  __syncthreads();
  if (threadIdx.x == 0) {
    float t = 0.f;
    for (int w = 0; w < 8; ++w) t += lpart[w];
    lossB[blockIdx.y * gridDim.x + blockIdx.x] = t;
  }
}

// SQ sim matrix [S,Q] via WMMA; one wave per 16x16 tile.
__global__ void k_sq_sim(const half_t* __restrict__ sechi, const half_t* __restrict__ seclo,
                         const half_t* __restrict__ quehi, const half_t* __restrict__ quelo,
                         float* __restrict__ simsq) {
  int lane = threadIdx.x, idx = lane & 15, h = lane >> 4;
  int n0 = blockIdx.x * 16;   // question (column)
  int s0 = blockIdx.y * 16;   // section (row)
  const half_t* ah = sechi + (size_t)(s0 + idx) * DDIM;
  const half_t* al = seclo + (size_t)(s0 + idx) * DDIM;
  const half_t* bh = quehi + (size_t)(n0 + idx) * DDIM;
  const half_t* bl = quelo + (size_t)(n0 + idx) * DDIM;
  v8f c = tile_gemm(ah, al, bh, bl, h);
#pragma unroll
  for (int r = 0; r < 8; ++r)
    simsq[(size_t)(s0 + r + 8 * h) * QDIM + (n0 + idx)] = c[r] / TAU_F;
}

__global__ void k_sq_rowsum(const float* __restrict__ simsq, const int* __restrict__ secq,
                            float* __restrict__ rowneg, float* __restrict__ msq) {
  int s = blockIdx.x, t = threadIdx.x;
  float a = 0.f, p = 0.f, cnt = 0.f;
  for (int q = t; q < QDIM; q += 256) {
    float sim = simsq[(size_t)s * QDIM + q];
    float e = expf(sim);
    a += e;
    if (secq[q] == s) { p += e; cnt += 1.f; }
  }
  a = block_reduce_sum(a);
  p = block_reduce_sum(p);
  cnt = block_reduce_sum(cnt);
  if (t == 0) {
    int M = (int)(cnt + 0.5f);
    bool valid = (M > 0) && ((QDIM - M) > 0);
    rowneg[s] = a - p;
    msq[s] = valid ? (float)M : 0.0f;
  }
}

__global__ void k_finalize(const float* __restrict__ lossB, const float* __restrict__ mval,
                           const float* __restrict__ simsq, const int* __restrict__ secq,
                           const float* __restrict__ rowneg, const float* __restrict__ msq,
                           float* __restrict__ out) {
  int t = threadIdx.x;
  float a = 0.f;
  for (int i = t; i < GRIDB; i += 256) a += lossB[i];
  float qs_sum = block_reduce_sum(a);
  float tq = 0.f;
  for (int q = t; q < QDIM; q += 256) tq += mval[q];
  tq = block_reduce_sum(tq);
  float sqa = 0.f;
  for (int q = t; q < QDIM; q += 256) {
    int s = secq[q];
    if (msq[s] > 0.0f) {
      float sim = simsq[(size_t)s * QDIM + q];
      sqa += logf(expf(sim) + rowneg[s]) - sim;
    }
  }
  sqa = block_reduce_sum(sqa);
  float tsq = (t < SDIM) ? msq[t] : 0.0f;
  tsq = block_reduce_sum(tsq);
  if (t == 0) {
    float ql = (tq > 0.f) ? qs_sum / tq : 0.f;
    float sl = (tsq > 0.f) ? sqa / tsq : 0.f;
    out[0] = ql + sl;
  }
}

extern "C" void kernel_launch(void* const* d_in, const int* in_sizes, int n_in,
                              void* d_out, int out_size, void* d_ws, size_t ws_size,
                              hipStream_t stream) {
  (void)in_sizes; (void)n_in; (void)out_size; (void)ws_size;
  const float* sec_in = (const float*)d_in[1];
  const float* que_in = (const float*)d_in[2];
  const float* sen_in = (const float*)d_in[3];
  const int* parq = (const int*)d_in[4];
  const int* secq = (const int*)d_in[5];
  const int* senp = (const int*)d_in[6];
  const int* pnum = (const int*)d_in[7];
  float* out = (float*)d_out;

  char* w = (char*)d_ws;
  size_t o = 0;
  auto alloc = [&](size_t bytes) -> char* {
    char* p = w + o; o += (bytes + 255) & ~(size_t)255; return p;
  };
  half_t* senhi = (half_t*)alloc((size_t)NDIM * DDIM * 2);
  half_t* senlo = (half_t*)alloc((size_t)NDIM * DDIM * 2);
  half_t* quehi = (half_t*)alloc((size_t)QDIM * DDIM * 2);
  half_t* quelo = (half_t*)alloc((size_t)QDIM * DDIM * 2);
  half_t* sechi = (half_t*)alloc((size_t)SDIM * DDIM * 2);
  half_t* seclo = (half_t*)alloc((size_t)SDIM * DDIM * 2);
  float* pa     = (float*)alloc((size_t)NSLICE * QDIM * 3 * 4);
  float* nng    = (float*)alloc((size_t)QDIM * 4);
  float* vflag  = (float*)alloc((size_t)QDIM * 4);
  float* mval   = (float*)alloc((size_t)QDIM * 4);
  float* lossB  = (float*)alloc((size_t)GRIDB * 4);
  float* simsq  = (float*)alloc((size_t)SDIM * QDIM * 4);
  float* rowneg = (float*)alloc((size_t)SDIM * 4);
  float* msq    = (float*)alloc((size_t)SDIM * 4);

  k_normalize_split<<<SDIM, DDIM, 0, stream>>>(sec_in, sechi, seclo);
  k_normalize_split<<<QDIM, DDIM, 0, stream>>>(que_in, quehi, quelo);
  k_normalize_split<<<NDIM, DDIM, 0, stream>>>(sen_in, senhi, senlo);

  dim3 gQS(QDIM / 128, NSLICE);
  k_pass_a<<<gQS, 256, 0, stream>>>(quehi, quelo, senhi, senlo, parq, senp, pa);
  k_g<<<QDIM / 256, 256, 0, stream>>>(pa, pnum, nng, vflag, mval);
  k_pass_b<<<gQS, 256, 0, stream>>>(quehi, quelo, senhi, senlo, parq, senp, nng, vflag, lossB);

  dim3 gSQ(QDIM / 16, SDIM / 16);
  k_sq_sim<<<gSQ, 32, 0, stream>>>(sechi, seclo, quehi, quelo, simsq);
  k_sq_rowsum<<<SDIM, 256, 0, stream>>>(simsq, secq, rowneg, msq);
  k_finalize<<<1, 256, 0, stream>>>(lossB, mval, simsq, secq, rowneg, msq, out);
}
